// decoder_57707180589239
// MI455X (gfx1250) — matrix operand order repaired and checked by reading, not compiled, not run on hardware
//
#include <hip/hip_runtime.h>
#include <hip/hip_bf16.h>

// ---------------------------------------------------------------------------
// CDNA5 (gfx1250): bf16 WMMA everywhere, async-to-LDS staging, hw tanh,
// software-pipelined k-loops (load k+1 fragments under k's WMMAs).
// Shapes: B=32, H=8, W=T=128, HD=HDDE=EMB=256, FEAT=512, HW=1024.
// ---------------------------------------------------------------------------

#define ASYNC_XT 1   // stage BiLSTM per-step Xih slab via global_load_async_to_lds

typedef __attribute__((ext_vector_type(16))) __bf16 v16bf;
typedef __attribute__((ext_vector_type(8)))  __bf16 v8bf;
typedef __attribute__((ext_vector_type(8)))  float  v8f;

union FragA { v16bf v; v8bf h[2]; };

static __device__ __forceinline__ float sigm(float x) {
    return 1.0f / (1.0f + __expf(-x));
}
static __device__ __forceinline__ float fast_tanh(float x) {
#if __has_builtin(__builtin_amdgcn_tanhf)
    return __builtin_amdgcn_tanhf(x);   // v_tanh_f32
#else
    return tanhf(x);
#endif
}
static __device__ __forceinline__ void wait_async0() {
#if __has_builtin(__builtin_amdgcn_s_wait_asynccnt)
    __builtin_amdgcn_s_wait_asynccnt(0);
#else
    asm volatile("s_wait_asynccnt 0x0" ::: "memory");
#endif
}

// A-matrix bf16 fragment (16x32), ISA 7.12.2: lane L row=base+(L&15);
// elems 0..7 -> K=k0+8*(L>>4)+j, elems 8..15 -> +16.
static __device__ __forceinline__ v16bf load_a_frag(const __bf16* base, int ld,
                                                    int row, int k0, int lane) {
    FragA f;
    const __bf16* p = base + (size_t)row * ld + k0 + ((lane >> 4) << 3);
    f.h[0] = *(const v8bf*)p;
    f.h[1] = *(const v8bf*)(p + 16);
    return f.v;
}
// B-matrix bf16 fragment (32x16) from W[N,K] row-major: lane L col=base+(L&15),
// elems 0..7 -> K = k0 + 8*(L>>4) + e, elems 8..15 -> +16.
static __device__ __forceinline__ v16bf load_b_frag(const __bf16* base, int ld,
                                                    int col, int k0, int lane) {
    FragA f;
    const __bf16* p = base + (size_t)col * ld + k0 + ((lane >> 4) << 3);
    f.h[0] = *(const v8bf*)p;
    f.h[1] = *(const v8bf*)(p + 16);
    return f.v;
}

#define WMMA_BF16(A_, B_, C_) \
    __builtin_amdgcn_wmma_f32_16x16x32_bf16(false, (A_), false, (B_), (short)0, (C_), false, false)

// ---------------------------------------------------------------------------
// Generic bf16 WMMA GEMM, 4-wide N register blocking (wave = 16x64 of C),
// software-pipelined: k+1 fragments load while k's WMMAs execute.
// C[m,n] = sum_k A[m,k]*W[n,k] + bias[n]; A:[M,K], W:[N,K] row-major.
// Output address: C + (m/row_div)*stride_hi + (m%row_div)*stride_lo + n.
// ---------------------------------------------------------------------------
__global__ __launch_bounds__(256)
void k_gemm(float* __restrict__ C, const __bf16* __restrict__ A,
            const __bf16* __restrict__ W, const float* __restrict__ bias,
            int M, int N, int K,
            int row_div, long long stride_hi, long long stride_lo) {
    int lane = threadIdx.x & 31;
    int wave = blockIdx.x * 8 + (threadIdx.x >> 5);
    int ngrp = N >> 6;
    int tm = (wave / ngrp) << 4;
    int tn = (wave % ngrp) << 6;
    if (tm >= M) return;
    int r = lane & 15, half = lane >> 4;
    v8f acc0 = {}, acc1 = {}, acc2 = {}, acc3 = {};
    v16bf a  = load_a_frag(A, K, tm + r, 0, lane);
    v16bf b0 = load_b_frag(W, K, tn + r,      0, lane);
    v16bf b1 = load_b_frag(W, K, tn + 16 + r, 0, lane);
    v16bf b2 = load_b_frag(W, K, tn + 32 + r, 0, lane);
    v16bf b3 = load_b_frag(W, K, tn + 48 + r, 0, lane);
    for (int k0 = 32; k0 < K; k0 += 32) {
        v16bf an  = load_a_frag(A, K, tm + r, k0, lane);
        v16bf bn0 = load_b_frag(W, K, tn + r,      k0, lane);
        v16bf bn1 = load_b_frag(W, K, tn + 16 + r, k0, lane);
        v16bf bn2 = load_b_frag(W, K, tn + 32 + r, k0, lane);
        v16bf bn3 = load_b_frag(W, K, tn + 48 + r, k0, lane);
        __builtin_prefetch((const void*)(A + (size_t)(tm + r) * K + k0 + 128), 0, 1);
        acc0 = WMMA_BF16(a, b0, acc0);
        acc1 = WMMA_BF16(a, b1, acc1);
        acc2 = WMMA_BF16(a, b2, acc2);
        acc3 = WMMA_BF16(a, b3, acc3);
        a = an; b0 = bn0; b1 = bn1; b2 = bn2; b3 = bn3;
    }
    acc0 = WMMA_BF16(a, b0, acc0);
    acc1 = WMMA_BF16(a, b1, acc1);
    acc2 = WMMA_BF16(a, b2, acc2);
    acc3 = WMMA_BF16(a, b3, acc3);
    v8f accs[4] = {acc0, acc1, acc2, acc3};
#pragma unroll
    for (int q = 0; q < 4; q++) {
        int col = tn + (q << 4) + r;
        float bv = bias ? bias[col] : 0.0f;
#pragma unroll
        for (int i = 0; i < 8; i++) {
            int m = tm + i + (half << 3);
            long long off = (long long)(m / row_div) * stride_hi +
                            (long long)(m % row_div) * stride_lo + col;
            C[off] = accs[q][i] + bv;
        }
    }
}

// ---------------------------------------------------------------------------
// conv_em implicit GEMM: M=32768 (b*1024+hw), N=256, K=4608; 4-wide N blocking.
// B loads for k+1 pipeline against the VALU im2col gather of A for k+1.
// ---------------------------------------------------------------------------
static __device__ __forceinline__ __bf16 conv_a_elem(const float* cfb, int y, int x, int k) {
    int ci = k / 9;
    int rem = k - ci * 9;
    int ky = rem / 3;
    int kx = rem - ky * 3;
    int yy = y + ky - 1, xx = x + kx - 1;
    float v = 0.0f;
    if ((unsigned)yy < 8u && (unsigned)xx < 128u)
        v = cfb[((size_t)ci * 8 + yy) * 128 + xx];
    return (__bf16)v;
}
static __device__ __forceinline__ v16bf conv_a_frag(const float* cfb, int y, int x,
                                                    int k0, int half) {
    FragA a;
    int kbA = k0 + (half << 3);
#pragma unroll
    for (int j = 0; j < 8; j++) {
        a.h[0][j] = conv_a_elem(cfb, y, x, kbA + j);
        a.h[1][j] = conv_a_elem(cfb, y, x, kbA + 16 + j);
    }
    return a.v;
}

__global__ __launch_bounds__(256)
void k_conv_em(float* __restrict__ x_em, const float* __restrict__ cf,
               const __bf16* __restrict__ wgt, const float* __restrict__ bias) {
    const int Ktot = 4608;
    int lane = threadIdx.x & 31;
    int wave = blockIdx.x * 8 + (threadIdx.x >> 5);
    int tm = (wave >> 2) << 4;     // 2048 m-tiles
    int tn = (wave & 3) << 6;      // 4 n-groups of 64
    int r = lane & 15, half = lane >> 4;
    int m = tm + r;
    int b = m >> 10, hw = m & 1023, y = hw >> 7, x = hw & 127;
    const float* cfb = cf + (size_t)b * 512 * 8 * 128;
    v8f acc0 = {}, acc1 = {}, acc2 = {}, acc3 = {};
    v16bf b0 = load_b_frag(wgt, Ktot, tn + r,      0, lane);
    v16bf b1 = load_b_frag(wgt, Ktot, tn + 16 + r, 0, lane);
    v16bf b2 = load_b_frag(wgt, Ktot, tn + 32 + r, 0, lane);
    v16bf b3 = load_b_frag(wgt, Ktot, tn + 48 + r, 0, lane);
    v16bf a  = conv_a_frag(cfb, y, x, 0, half);
    for (int k0 = 32; k0 < Ktot; k0 += 32) {
        v16bf bn0 = load_b_frag(wgt, Ktot, tn + r,      k0, lane);
        v16bf bn1 = load_b_frag(wgt, Ktot, tn + 16 + r, k0, lane);
        v16bf bn2 = load_b_frag(wgt, Ktot, tn + 32 + r, k0, lane);
        v16bf bn3 = load_b_frag(wgt, Ktot, tn + 48 + r, k0, lane);
        acc0 = WMMA_BF16(a, b0, acc0);
        acc1 = WMMA_BF16(a, b1, acc1);
        acc2 = WMMA_BF16(a, b2, acc2);
        acc3 = WMMA_BF16(a, b3, acc3);
        a = conv_a_frag(cfb, y, x, k0, half);
        b0 = bn0; b1 = bn1; b2 = bn2; b3 = bn3;
    }
    acc0 = WMMA_BF16(a, b0, acc0);
    acc1 = WMMA_BF16(a, b1, acc1);
    acc2 = WMMA_BF16(a, b2, acc2);
    acc3 = WMMA_BF16(a, b3, acc3);
    v8f accs[4] = {acc0, acc1, acc2, acc3};
#pragma unroll
    for (int q = 0; q < 4; q++) {
        int col = tn + (q << 4) + r;
        float bv = bias[col];
#pragma unroll
        for (int i = 0; i < 8; i++) {
            int mm = tm + i + (half << 3);
            x_em[(size_t)mm * 256 + col] = accs[q][i] + bv;
        }
    }
}

// ---------------------------------------------------------------------------
// Shared recurrent-GEMM body: gates[32,1024] = Asm[32,K] @ W^T + adds.
// 32 groups (2 m-tiles x 16 n-groups of 64); 8 waves -> 4 groups each.
// ---------------------------------------------------------------------------
static __device__ __forceinline__
void gemm_gates(const __bf16* __restrict__ Asm, int K, const __bf16* __restrict__ W,
                float* __restrict__ gates, const float* __restrict__ add_vec,
                const float* __restrict__ add_mat, int lane, int wave) {
    int r = lane & 15, half = lane >> 4;
    for (int grp = wave; grp < 32; grp += 8) {
        int tm = (grp >> 4) << 4, tn = (grp & 15) << 6;
        v8f acc0 = {}, acc1 = {}, acc2 = {}, acc3 = {};
        v16bf a  = load_a_frag(Asm, K, tm + r, 0, lane);
        v16bf b0 = load_b_frag(W, K, tn + r,      0, lane);
        v16bf b1 = load_b_frag(W, K, tn + 16 + r, 0, lane);
        v16bf b2 = load_b_frag(W, K, tn + 32 + r, 0, lane);
        v16bf b3 = load_b_frag(W, K, tn + 48 + r, 0, lane);
        for (int k0 = 32; k0 < K; k0 += 32) {
            v16bf an  = load_a_frag(Asm, K, tm + r, k0, lane);
            v16bf bn0 = load_b_frag(W, K, tn + r,      k0, lane);
            v16bf bn1 = load_b_frag(W, K, tn + 16 + r, k0, lane);
            v16bf bn2 = load_b_frag(W, K, tn + 32 + r, k0, lane);
            v16bf bn3 = load_b_frag(W, K, tn + 48 + r, k0, lane);
            acc0 = WMMA_BF16(a, b0, acc0);
            acc1 = WMMA_BF16(a, b1, acc1);
            acc2 = WMMA_BF16(a, b2, acc2);
            acc3 = WMMA_BF16(a, b3, acc3);
            a = an; b0 = bn0; b1 = bn1; b2 = bn2; b3 = bn3;
        }
        acc0 = WMMA_BF16(a, b0, acc0);
        acc1 = WMMA_BF16(a, b1, acc1);
        acc2 = WMMA_BF16(a, b2, acc2);
        acc3 = WMMA_BF16(a, b3, acc3);
        v8f accs[4] = {acc0, acc1, acc2, acc3};
#pragma unroll
        for (int q = 0; q < 4; q++) {
            int col = tn + (q << 4) + r;
            float add = add_vec ? add_vec[col] : 0.0f;
#pragma unroll
            for (int i = 0; i < 8; i++) {
                int mm = tm + i + (half << 3);
                float v = accs[q][i] + add;
                if (add_mat) v += add_mat[(size_t)mm * 1024 + col];
                gates[mm * 1024 + col] = v;
            }
        }
    }
}

// ---------------------------------------------------------------------------
// Persistent BiLSTM, grid=2 (dir). LDS: h(bf16) 16K + c 32K + gates 128K
// + Xt slab 128K = 304KB (CDNA5 WGP has 320KB). Per-step Xih slab staged with
// global_load_async_to_lds_b128 (ASYNCcnt). Writes h bf16 into Hcat[T*32,512].
// ---------------------------------------------------------------------------
__global__ __launch_bounds__(256)
void k_bilstm(const float* __restrict__ XihF, const float* __restrict__ XihB,
              const __bf16* __restrict__ WhF, const __bf16* __restrict__ WhB,
              __bf16* __restrict__ Hcat) {
    int dir = blockIdx.x;
    const float* Xih = dir ? XihB : XihF;
    const __bf16* W = dir ? WhB : WhF;
    extern __shared__ char smem[];
    __bf16* hA    = (__bf16*)smem;                                   // [32,256]
    float*  cst   = (float*)(smem + 16384);                          // [32,256]
    float*  gates = (float*)(smem + 16384 + 32768);                  // [32,1024]
#if ASYNC_XT
    float*  XtL   = (float*)(smem + 16384 + 32768 + 131072);         // [32,1024]
#endif
    int tid = threadIdx.x;
    for (int i = tid; i < 32 * 256; i += 256) { hA[i] = (__bf16)0.0f; cst[i] = 0.0f; }
    __syncthreads();
    int lane = tid & 31, wave = tid >> 5;
    for (int s = 0; s < 128; s++) {
        int t = dir ? (127 - s) : s;
        const float* Xt = Xih + (size_t)t * 32 * 1024;
#if ASYNC_XT
        // async DMA the 128KB gate-bias slab for this step into LDS
        for (int blk = wave; blk < 256; blk += 8) {         // 512B per instruction
            int elt = blk * 128 + lane * 4;
            unsigned loff = (unsigned)(size_t)(XtL + elt);
            const float* g = Xt + elt;
            asm volatile("global_load_async_to_lds_b128 %0, %1, off"
                         :: "v"(loff), "v"(g) : "memory");
        }
        wait_async0();
        __syncthreads();
        const float* Xadd = XtL;
#else
        const float* Xadd = Xt;
#endif
        gemm_gates(hA, 256, W, gates, nullptr, Xadd, lane, wave);
        __syncthreads();
        for (int i = tid; i < 32 * 256; i += 256) {
            int bb = i >> 8, j = i & 255;
            float gi = gates[bb * 1024 + j];
            float gf = gates[bb * 1024 + 256 + j];
            float gg = gates[bb * 1024 + 512 + j];
            float go = gates[bb * 1024 + 768 + j];
            float cv = sigm(gf) * cst[i] + sigm(gi) * fast_tanh(gg);
            float hv = sigm(go) * fast_tanh(cv);
            cst[i] = cv;
            hA[i] = (__bf16)hv;
            Hcat[((size_t)t * 32 + bb) * 512 + dir * 256 + j] = (__bf16)hv;
        }
        __syncthreads();
    }
}

// ---------------------------------------------------------------------------
// Fused decoder step: cell0 then cell1 in one block (states in global fp32).
// cell1 folds x@W_ih and h@W_hh into one K=512 GEMM via A=[h0_new, h1_old].
// Writes new h1 to attn_out[:, t, 0:256].
// ---------------------------------------------------------------------------
__global__ __launch_bounds__(256)
void k_dec_cells(float* __restrict__ h0, float* __restrict__ c0,
                 float* __restrict__ h1, float* __restrict__ c1,
                 const __bf16* __restrict__ W0, const __bf16* __restrict__ Wcat,
                 const float* __restrict__ add_vec0, const float* __restrict__ add_mat0,
                 const float* __restrict__ c1_bias,
                 float* __restrict__ attn_out, long long bstride) {
    extern __shared__ char smem[];
    __bf16* Asm   = (__bf16*)smem;                 // [32,512]
    float*  gates = (float*)(smem + 32 * 512 * 2); // [32,1024]
    int tid = threadIdx.x, lane = tid & 31, wave = tid >> 5;
    for (int i = tid; i < 8192; i += 256) {
        int bb = i >> 8, j = i & 255;
        Asm[bb * 256 + j] = (__bf16)h0[i];
    }
    __syncthreads();
    gemm_gates(Asm, 256, W0, gates, add_vec0, add_mat0, lane, wave);
    __syncthreads();
    for (int i = tid; i < 8192; i += 256) {
        int bb = i >> 8, j = i & 255;
        float gi = gates[bb * 1024 + j];
        float gf = gates[bb * 1024 + 256 + j];
        float gg = gates[bb * 1024 + 512 + j];
        float go = gates[bb * 1024 + 768 + j];
        float cv = sigm(gf) * c0[i] + sigm(gi) * fast_tanh(gg);
        float hv = sigm(go) * fast_tanh(cv);
        c0[i] = cv;
        h0[i] = hv;
        Asm[bb * 512 + j]       = (__bf16)hv;       // cell1 A = [h0_new, h1_old]
        Asm[bb * 512 + 256 + j] = (__bf16)h1[i];
    }
    __syncthreads();
    gemm_gates(Asm, 512, Wcat, gates, c1_bias, nullptr, lane, wave);
    __syncthreads();
    for (int i = tid; i < 8192; i += 256) {
        int bb = i >> 8, j = i & 255;
        float gi = gates[bb * 1024 + j];
        float gf = gates[bb * 1024 + 256 + j];
        float gg = gates[bb * 1024 + 512 + j];
        float go = gates[bb * 1024 + 768 + j];
        float cv = sigm(gf) * c1[i] + sigm(gi) * fast_tanh(gg);
        float hv = sigm(go) * fast_tanh(cv);
        c1[i] = cv;
        h1[i] = hv;
        attn_out[(long long)bb * bstride + j] = hv;
    }
}

// ---------------------------------------------------------------------------
// Additive attention core (one query row per block, 256 threads).
// s[hw]=att_w.tanh(x_em[b,hw,:]+g)+att_b; softmax over 1024; ctx = a @ x_em[b].
// ---------------------------------------------------------------------------
static __device__ __forceinline__
void attention_core(const float* __restrict__ xb, float* __restrict__ outp,
                    const float* g, const float* aw, float ab,
                    float* s, float* red, int tid) {
    for (int p = tid; p < 1024; p += 256) {
        const float* xr = xb + (size_t)p * 256;
        float acc = 0.0f;
        for (int e = 0; e < 256; e++) acc += aw[e] * fast_tanh(xr[e] + g[e]);
        s[p] = acc + ab;
    }
    __syncthreads();
    float mx = -1e30f;
    for (int p = tid; p < 1024; p += 256) mx = fmaxf(mx, s[p]);
    red[tid] = mx;
    __syncthreads();
    for (int st = 128; st > 0; st >>= 1) {
        if (tid < st) red[tid] = fmaxf(red[tid], red[tid + st]);
        __syncthreads();
    }
    mx = red[0];
    __syncthreads();
    float sum = 0.0f;
    for (int p = tid; p < 1024; p += 256) {
        float e0 = __expf(s[p] - mx);
        s[p] = e0;
        sum += e0;
    }
    red[tid] = sum;
    __syncthreads();
    for (int st = 128; st > 0; st >>= 1) {
        if (tid < st) red[tid] += red[tid + st];
        __syncthreads();
    }
    float inv = 1.0f / red[0];
    float acc = 0.0f;
    for (int p = 0; p < 1024; p++) acc += s[p] * xb[(size_t)p * 256 + tid];
    outp[tid] = acc * inv;
}

// CTC attention: r = blockIdx.x = t*32+b, G precomputed by WMMA GEMM.
__global__ __launch_bounds__(256)
void k_attention(float* __restrict__ out, long long out_bstride, long long out_tstride,
                 const float* __restrict__ G, const float* __restrict__ x_em,
                 const float* __restrict__ att_w, const float* __restrict__ att_b,
                 int bmod) {
    __shared__ float g[256], aw[256], s[1024], red[256];
    int tid = threadIdx.x, rr = blockIdx.x;
    int b = rr % bmod, t = rr / bmod;
    g[tid] = G[(size_t)rr * 256 + tid];
    aw[tid] = att_w[tid];
    __syncthreads();
    attention_core(x_em + (size_t)b * 262144,
                   out + (long long)b * out_bstride + (long long)t * out_tstride,
                   g, aw, att_b[0], s, red, tid);
}

// Decoder attention: block b; fuses g = h1[b]@conv_h_w^T + conv_h_b (64K MACs).
__global__ __launch_bounds__(256)
void k_dec_attn(float* __restrict__ out, long long out_bstride,
                const float* __restrict__ h1, const float* __restrict__ chw,
                const float* __restrict__ chb, const float* __restrict__ x_em,
                const float* __restrict__ att_w, const float* __restrict__ att_b) {
    __shared__ float h1s[256], g[256], aw[256], s[1024], red[256];
    int tid = threadIdx.x, b = blockIdx.x;
    h1s[tid] = h1[b * 256 + tid];
    aw[tid] = att_w[tid];
    __syncthreads();
    float acc = chb[tid];
    const float* wr = chw + (size_t)tid * 256;
    for (int k = 0; k < 256; k++) acc += h1s[k] * wr[k];
    g[tid] = acc;
    __syncthreads();
    attention_core(x_em + (size_t)b * 262144, out + (long long)b * out_bstride,
                   g, aw, att_b[0], s, red, tid);
}

// ------------------------------ small utils --------------------------------
__global__ void k_cvt_bf16(__bf16* dst, const float* src, int n) {
    int i = blockIdx.x * 256 + threadIdx.x;
    if (i < n) dst[i] = (__bf16)src[i];
}
__global__ void k_zero_f32(float* p, int n) {
    int i = blockIdx.x * 256 + threadIdx.x;
    if (i < n) p[i] = 0.0f;
}
__global__ void k_vec_ih(float* out, const float* e, const float* wih, const float* bias) {
    int n = blockIdx.x * 256 + threadIdx.x;
    if (n >= 1024) return;
    float acc = bias[n];
    const float* wr = wih + (size_t)n * 256;
    for (int k = 0; k < 256; k++) acc += e[k] * wr[k];
    out[n] = acc;
}
__global__ void k_build_wcat(__bf16* out, const float* wih, const float* whh) {
    int i = blockIdx.x * 256 + threadIdx.x;
    if (i >= 1024 * 512) return;
    int n = i >> 9, k = i & 511;
    float v = (k < 256) ? wih[n * 256 + k] : whh[n * 256 + k - 256];
    out[i] = (__bf16)v;
}
__global__ void k_scatter_bt(float* out, const float* in) {
    int rr = blockIdx.x, j = threadIdx.x;
    int b = rr & 31, t = rr >> 5;
    out[(size_t)b * 128 * 512 + (size_t)t * 512 + j] = in[(size_t)rr * 256 + j];
}

// ---------------------------------------------------------------------------
extern "C" void kernel_launch(void* const* d_in, const int* in_sizes, int n_in,
                              void* d_out, int out_size, void* d_ws, size_t ws_size,
                              hipStream_t stream) {
    (void)in_sizes; (void)n_in; (void)out_size; (void)ws_size;
    const float* hidden_en = (const float*)d_in[0];
    const float* conv_f    = (const float*)d_in[1];
    const float* conv_em_w = (const float*)d_in[2];
    const float* conv_em_b = (const float*)d_in[3];
    const float* w_ih_f = (const float*)d_in[4];
    const float* w_hh_f = (const float*)d_in[5];
    const float* b_f    = (const float*)d_in[6];
    const float* w_ih_b = (const float*)d_in[7];
    const float* w_hh_b = (const float*)d_in[8];
    const float* b_b    = (const float*)d_in[9];
    const float* lin_w  = (const float*)d_in[10];
    const float* lin_b  = (const float*)d_in[11];
    const float* conv_h_w = (const float*)d_in[12];
    const float* conv_h_b = (const float*)d_in[13];
    const float* att_w  = (const float*)d_in[14];
    const float* att_b  = (const float*)d_in[15];
    const float* c0_wih = (const float*)d_in[16];
    const float* c0_whh = (const float*)d_in[17];
    const float* c0_b   = (const float*)d_in[18];
    const float* c1_wih = (const float*)d_in[19];
    const float* c1_whh = (const float*)d_in[20];
    const float* c1_b   = (const float*)d_in[21];
    const float* embed_w = (const float*)d_in[22];

    float* ctc  = (float*)d_out;                 // [32,128,512]
    float* attn = ctc + (size_t)32 * 128 * 512;  // [32,128,512]

    char* ws = (char*)d_ws;
    size_t cur = 0;
    auto alloc = [&](size_t bytes) -> char* {
        char* p = ws + cur;
        cur += (bytes + 255) & ~(size_t)255;
        return p;
    };

    float*  x_em      = (float*)alloc((size_t)32 * 1024 * 256 * 4);
    __bf16* he_bf     = (__bf16*)alloc((size_t)4096 * 256 * 2);
    __bf16* cwem_bf   = (__bf16*)alloc((size_t)256 * 4608 * 2);
    __bf16* wihf_bf   = (__bf16*)alloc((size_t)1024 * 256 * 2);
    __bf16* whhf_bf   = (__bf16*)alloc((size_t)1024 * 256 * 2);
    __bf16* wihb_bf   = (__bf16*)alloc((size_t)1024 * 256 * 2);
    __bf16* whhb_bf   = (__bf16*)alloc((size_t)1024 * 256 * 2);
    __bf16* linw_bf   = (__bf16*)alloc((size_t)256 * 512 * 2);
    __bf16* chw_bf    = (__bf16*)alloc((size_t)256 * 256 * 2);
    __bf16* c0wih_bf  = (__bf16*)alloc((size_t)1024 * 256 * 2);
    __bf16* c0whh_bf  = (__bf16*)alloc((size_t)1024 * 256 * 2);
    __bf16* wcat_bf   = (__bf16*)alloc((size_t)1024 * 512 * 2);
    float*  Xih_f     = (float*)alloc((size_t)4096 * 1024 * 4);
    float*  Xih_b     = (float*)alloc((size_t)4096 * 1024 * 4);
    __bf16* Hcat      = (__bf16*)alloc((size_t)4096 * 512 * 2);
    float*  hidden_de = (float*)alloc((size_t)4096 * 256 * 4);
    __bf16* hd_bf     = (__bf16*)alloc((size_t)4096 * 256 * 2);
    float*  G_ctc     = (float*)alloc((size_t)4096 * 256 * 4);
    float*  Xih0      = (float*)alloc((size_t)32 * 1024 * 4);
    float*  vec0      = (float*)alloc((size_t)1024 * 4);
    float*  states    = (float*)alloc((size_t)4 * 32 * 256 * 4);
    float* h0 = states, *cc0 = states + 8192, *h1 = states + 16384, *cc1 = states + 24576;

    auto cvt = [&](__bf16* dst, const float* src, int n) {
        k_cvt_bf16<<<(n + 255) / 256, 256, 0, stream>>>(dst, src, n);
    };
    auto gblocks = [](long long waves) { return (int)((waves + 7) / 8); };

    // --- conversions ---
    cvt(he_bf, hidden_en, 4096 * 256);
    cvt(cwem_bf, conv_em_w, 256 * 4608);
    cvt(wihf_bf, w_ih_f, 1024 * 256);
    cvt(whhf_bf, w_hh_f, 1024 * 256);
    cvt(wihb_bf, w_ih_b, 1024 * 256);
    cvt(whhb_bf, w_hh_b, 1024 * 256);
    cvt(linw_bf, lin_w, 256 * 512);
    cvt(chw_bf, conv_h_w, 256 * 256);
    cvt(c0wih_bf, c0_wih, 1024 * 256);
    cvt(c0whh_bf, c0_whh, 1024 * 256);
    k_build_wcat<<<(1024 * 512 + 255) / 256, 256, 0, stream>>>(wcat_bf, c1_wih, c1_whh);

    // --- conv_em implicit GEMM: waves = 2048 m-tiles * 4 n-groups ---
    k_conv_em<<<1024, 256, 0, stream>>>(x_em, conv_f, cwem_bf, conv_em_b);

    // --- batched x@W_ih for both LSTM directions (bias folded in) ---
    k_gemm<<<gblocks(256LL * 16), 256, 0, stream>>>(Xih_f, he_bf, wihf_bf, b_f,
                                                    4096, 1024, 256, 1, 1024, 0);
    k_gemm<<<gblocks(256LL * 16), 256, 0, stream>>>(Xih_b, he_bf, wihb_bf, b_b,
                                                    4096, 1024, 256, 1, 1024, 0);

    // --- recurrent BiLSTM (persistent; async-to-LDS staged Xih slabs) ---
#if ASYNC_XT
    const size_t bilstm_smem = 16384 + 32768 + 131072 + 131072;   // 304KB
#else
    const size_t bilstm_smem = 16384 + 32768 + 131072;
#endif
    k_bilstm<<<2, 256, bilstm_smem, stream>>>(Xih_f, Xih_b, whhf_bf, whhb_bf, Hcat);

    // --- hidden_de = Hcat @ lin_w^T + lin_b ---
    k_gemm<<<gblocks(256LL * 4), 256, 0, stream>>>(hidden_de, Hcat, linw_bf, lin_b,
                                                   4096, 256, 512, 1, 256, 0);
    cvt(hd_bf, hidden_de, 4096 * 256);

    // --- CTC path ---
    k_gemm<<<gblocks(256LL * 4), 256, 0, stream>>>(G_ctc, hd_bf, chw_bf, conv_h_b,
                                                   4096, 256, 256, 1, 256, 0);
    k_scatter_bt<<<4096, 256, 0, stream>>>(ctc, hidden_de);
    k_attention<<<4096, 256, 0, stream>>>(ctc + 256, (long long)128 * 512, 512,
                                          G_ctc, x_em, att_w, att_b, 32);

    // --- decoder setup ---
    k_gemm<<<gblocks(2LL * 16), 256, 0, stream>>>(Xih0, he_bf + (size_t)127 * 32 * 256,
                                                  c0wih_bf, c0_b, 32, 1024, 256, 1, 1024, 0);
    k_vec_ih<<<4, 256, 0, stream>>>(vec0, embed_w, c0_wih, c0_b);
    k_zero_f32<<<(4 * 32 * 256 + 255) / 256, 256, 0, stream>>>(states, 4 * 32 * 256);

    // --- decoder: 128 steps x (fused cells, fused proj+attention) ---
    const size_t cell_smem = 32 * 512 * 2 + 32 * 1024 * 4;  // 160KB
    for (int t = 0; t < 128; t++) {
        k_dec_cells<<<1, 256, cell_smem, stream>>>(
            h0, cc0, h1, cc1, c0whh_bf, wcat_bf,
            (t == 0) ? nullptr : vec0, (t == 0) ? Xih0 : nullptr, c1_b,
            attn + (size_t)t * 512, (long long)128 * 512);
        k_dec_attn<<<32, 256, 0, stream>>>(attn + (size_t)t * 512 + 256,
                                           (long long)128 * 512,
                                           h1, conv_h_w, conv_h_b, x_em, att_w, att_b);
    }
}
